// Encoder_16767552323943
// MI455X (gfx1250) — matrix operand order repaired and checked by reading, not compiled, not run on hardware
//
#include <hip/hip_runtime.h>
#include <hip/hip_bf16.h>
#include <math.h>
#include <stdint.h>

typedef __attribute__((ext_vector_type(16))) _Float16 v16h;
typedef __attribute__((ext_vector_type(8)))  _Float16 v8h;
typedef __attribute__((ext_vector_type(8)))  float    v8f;

union Frag16 { v16h v; v8h h[2]; };

#define DIMSZ 768
#define HIDSZ 1536
#define NBATCH 8
#define NSEQ 1024
#define NHEAD 16
#define HDIM 48
#define HDPAD 64
#define ROWS (NBATCH*NSEQ)   /* 8192 */

#if defined(__has_builtin)
#  if __has_builtin(__builtin_amdgcn_global_load_async_to_lds_b128)
#    define HAS_ASYNC_LDS 1
#  endif
#  if __has_builtin(__builtin_amdgcn_s_wait_asynccnt)
#    define HAS_WAIT_ASYNC 1
#  endif
#endif

// Builtin signature (from hipcc diagnostic): first arg is a non-const
// addrspace(1) pointer to a 16-byte int vector, second is the addrspace(3)
// LDS-side pointer, then imm offset + imm cpol.
typedef int int4v __attribute__((vector_size(16)));
typedef __attribute__((address_space(1))) int4v* gb128_t;
typedef __attribute__((address_space(3))) int4v* lb128_t;

// Async 16B global -> LDS copy (ASYNCcnt-tracked). Low 32 bits of a generic
// LDS address are the LDS byte offset on gfx1250 (ISA 10.2 aperture rules),
// so the integer-truncation cast to as(3) is semantically correct.
__device__ __forceinline__ void async_copy_b128(const void* g, void* l) {
#ifdef HAS_ASYNC_LDS
  __builtin_amdgcn_global_load_async_to_lds_b128(
      (gb128_t)(uintptr_t)g, (lb128_t)(uint32_t)(uintptr_t)l, 0, 0);
#else
  (void)g; (void)l;
#endif
}

__device__ __forceinline__ void wait_async0() {
#ifdef HAS_WAIT_ASYNC
  __builtin_amdgcn_s_wait_asynccnt(0);
#elif defined(HAS_ASYNC_LDS)
  asm volatile("s_wait_asynccnt 0x0" ::: "memory");
#endif
}

__device__ __forceinline__ v8f wmma16(const Frag16& a, const Frag16& b, v8f c) {
  return __builtin_amdgcn_wmma_f32_16x16x32_f16(
      /*neg_a=*/false, a.v, /*neg_b=*/false, b.v,
      /*c_mod=*/(short)0, c, /*reuse_a=*/false, /*reuse_b=*/false);
}

// ---------------------------------------------------------------- convert
__global__ __launch_bounds__(256)
void cvt_f16(const float* __restrict__ in, _Float16* __restrict__ out, int n) {
  int i = blockIdx.x * 256 + threadIdx.x;
  if (i < n) out[i] = (_Float16)in[i];
}

// ---------------------------------------------------------------- GEMM
// C[M,N] = A[M,K] @ B[K,N]   (A,B f16 row-major, accum f32)
// Workgroup: 256 threads = 8 waves. WG tile 128x128. Wave w owns rows
// [w*16, w*16+16) of the tile, all 128 cols (8 accumulator tiles).
// MODE 0: out f32 = C + bias
// MODE 1: out f16 = gelu_exact(C + bias)
// MODE 2: qkv scatter into padded [B,H,S,64] f16 buffers, q scaled.
#define GLDB 40   // padded LDS stride (halfs) for B tile
template<int MODE>
__global__ __launch_bounds__(256)
void gemm_wmma(const _Float16* __restrict__ A, const _Float16* __restrict__ Bm,
               const float* __restrict__ bias, int M, int N, int K,
               float* __restrict__ outF, _Float16* __restrict__ outH,
               _Float16* __restrict__ qb, _Float16* __restrict__ kb,
               _Float16* __restrict__ vb, float qscale)
{
  __shared__ _Float16 Bs[128 * GLDB];   // transposed: Bs[n][kk]
  const int tid   = threadIdx.x;
  const int wave  = tid >> 5;
  const int lane  = tid & 31;
  const int lhalf = lane >> 4;          // 0 / 1
  const int l16   = lane & 15;
  const int nblk  = N >> 7;
  const int bm = blockIdx.x / nblk, bn = blockIdx.x % nblk;
  const int m0 = bm << 7, n0 = bn << 7;
  const int arow = m0 + wave * 16 + l16;

  v8f acc[8] = {};

  for (int k0 = 0; k0 < K; k0 += 32) {
    // ---- stage B[32 x 128] transposed into LDS (512 8-half segments)
    #pragma unroll
    for (int r = 0; r < 2; ++r) {
      int idx = tid + 256 * r;          // 0..511
      int kk  = idx >> 4;               // 0..31
      int seg = idx & 15;               // 0..15
      const _Float16* src = Bm + (size_t)(k0 + kk) * N + n0 + seg * 8;
      v8h bv = *(const v8h*)src;
      if (k0 + 32 < K) __builtin_prefetch(src + (size_t)32 * N, 0, 3);
      int nb = seg * 8;
      #pragma unroll
      for (int e = 0; e < 8; ++e) Bs[(nb + e) * GLDB + kk] = bv[e];
    }
    __syncthreads();

    // ---- A fragment (16x32), direct from global
    Frag16 af;
    const _Float16* ap = A + (size_t)arow * K + k0 + lhalf * 8;
    af.h[0] = *(const v8h*)(ap);
    af.h[1] = *(const v8h*)(ap + 16);
    if (k0 + 32 < K) __builtin_prefetch(ap + 32, 0, 3);

    #pragma unroll
    for (int nt = 0; nt < 8; ++nt) {
      Frag16 bf;
      const _Float16* bp = &Bs[(nt * 16 + l16) * GLDB + lhalf * 8];
      bf.h[0] = *(const v8h*)bp;
      bf.h[1] = *(const v8h*)(bp + 16);
      acc[nt] = wmma16(af, bf, acc[nt]);
    }
    __syncthreads();
  }

  // ---- epilogue  (C layout: row = i + 8*lhalf, col = l16 within tile)
  const int rbase = m0 + wave * 16 + 8 * lhalf;
  #pragma unroll
  for (int nt = 0; nt < 8; ++nt) {
    int col = n0 + nt * 16 + l16;
    float bb = (MODE == 2) ? 0.0f : bias[col];
    #pragma unroll
    for (int i = 0; i < 8; ++i) {
      int row = rbase + i;
      float val = acc[nt][i] + bb;
      if (MODE == 0) {
        outF[(size_t)row * N + col] = val;
      } else if (MODE == 1) {
        float g = 0.5f * val * (1.0f + erff(val * 0.70710678118654752f));
        outH[(size_t)row * N + col] = (_Float16)g;
      } else {
        int which = col / DIMSZ;
        int d = col - which * DIMSZ;
        int head = d / HDIM;
        int hd = d - head * HDIM;
        int b = row >> 10, s = row & 1023;
        float sv = (which == 0) ? val * qscale : val;
        _Float16* dst = (which == 0) ? qb : ((which == 1) ? kb : vb);
        dst[(((size_t)(b * NHEAD + head) * NSEQ + s) << 6) + hd] = (_Float16)sv;
      }
    }
  }
}

// ---------------------------------------------------------------- attention
// Grid: (B*H) * (S/64) blocks, 128 threads (4 waves). Each wave: 16 q rows.
// Flash-attention: stream K/V in 32-key chunks through LDS, online softmax.
// K chunk is a verbatim row-major copy -> async global->LDS path (ASYNCcnt).
#define KLD 72   // Ks stride (halfs)
#define VLD 40   // Vt / Ps stride (halfs)
__global__ __launch_bounds__(128)
void attn_wmma(const _Float16* __restrict__ qh, const _Float16* __restrict__ kh,
               const _Float16* __restrict__ vh, _Float16* __restrict__ outh)
{
  __shared__ _Float16 Ks[32 * KLD];        // [key][hd]  (row-major chunk)
  __shared__ _Float16 Vt[64 * VLD];        // [hd][key]  (transposed chunk)
  __shared__ _Float16 Ps[4 * 16 * VLD];    // per-wave P tile [m][key]

  const int tid   = threadIdx.x;
  const int wave  = tid >> 5;
  const int lane  = tid & 31;
  const int lhalf = lane >> 4;
  const int l16   = lane & 15;

  const int bh   = blockIdx.x >> 4;        // 16 row-blocks per (b,h)
  const int sblk = blockIdx.x & 15;
  const size_t base = (size_t)bh * NSEQ * HDPAD;
  const int qrow0 = sblk * 64 + wave * 16;

  // resident Q fragments (16 rows x 64 hd = 2 K-steps)
  Frag16 qf[2];
  {
    const _Float16* qp = qh + base + (size_t)(qrow0 + l16) * HDPAD;
    #pragma unroll
    for (int ks = 0; ks < 2; ++ks) {
      const _Float16* p = qp + ks * 32 + lhalf * 8;
      qf[ks].h[0] = *(const v8h*)p;
      qf[ks].h[1] = *(const v8h*)(p + 16);
    }
  }

  v8f oacc[4] = {};
  float rm[8], rs[8];
  #pragma unroll
  for (int i = 0; i < 8; ++i) { rm[i] = -1e30f; rs[i] = 0.0f; }
  _Float16* Pw = &Ps[wave * 16 * VLD];

  for (int kc = 0; kc < NSEQ; kc += 32) {
    // ---- stage K (row-major, async-to-LDS) and V (transposed, sync)
    {
      int key = tid >> 2;                  // 0..31
      int seg = tid & 3;                   // 0..3 (16 halfs each)
      const _Float16* kp = kh + base + (size_t)(kc + key) * HDPAD + seg * 16;
      _Float16* kd = &Ks[key * KLD + seg * 16];
#ifdef HAS_ASYNC_LDS
      async_copy_b128(kp, kd);
      async_copy_b128(kp + 8, kd + 8);
#else
      v8h a0 = *(const v8h*)kp, a1 = *(const v8h*)(kp + 8);
      *(v8h*)kd = a0; *(v8h*)(kd + 8) = a1;
#endif
      const _Float16* vp = vh + base + (size_t)(kc + key) * HDPAD + seg * 16;
      v8h b0 = *(const v8h*)vp, b1 = *(const v8h*)(vp + 8);
      #pragma unroll
      for (int e = 0; e < 8; ++e) Vt[(seg * 16 + e) * VLD + key] = b0[e];
      #pragma unroll
      for (int e = 0; e < 8; ++e) Vt[(seg * 16 + 8 + e) * VLD + key] = b1[e];
    }
    wait_async0();
    __syncthreads();

    // ---- S = Q @ K^T   (16 x 32, two 16x16 tiles)
    v8f sacc[2] = {};
    #pragma unroll
    for (int nt = 0; nt < 2; ++nt) {
      #pragma unroll
      for (int ks = 0; ks < 2; ++ks) {
        Frag16 bf;
        const _Float16* bp = &Ks[(nt * 16 + l16) * KLD + ks * 32 + lhalf * 8];
        bf.h[0] = *(const v8h*)bp;
        bf.h[1] = *(const v8h*)(bp + 16);
        sacc[nt] = wmma16(qf[ks], bf, sacc[nt]);
      }
    }

    // ---- online softmax (row reductions across 16-lane half-waves)
    #pragma unroll
    for (int i = 0; i < 8; ++i) {
      float s0 = sacc[0][i], s1 = sacc[1][i];
      float mx = fmaxf(s0, s1);
      #pragma unroll
      for (int o = 8; o > 0; o >>= 1) mx = fmaxf(mx, __shfl_xor(mx, o, 16));
      float nm = fmaxf(rm[i], mx);
      float corr = __expf(rm[i] - nm);
      rm[i] = nm;
      float p0 = __expf(s0 - nm), p1 = __expf(s1 - nm);
      float ps = p0 + p1;
      #pragma unroll
      for (int o = 8; o > 0; o >>= 1) ps += __shfl_xor(ps, o, 16);
      rs[i] = rs[i] * corr + ps;
      #pragma unroll
      for (int nt = 0; nt < 4; ++nt) oacc[nt][i] = oacc[nt][i] * corr;
      int m = i + 8 * lhalf;
      Pw[m * VLD + l16]      = (_Float16)p0;
      Pw[m * VLD + 16 + l16] = (_Float16)p1;
    }

    // ---- O += P @ V    (A-frag for P via LDS round-trip)
    Frag16 pf;
    const _Float16* pp = &Pw[l16 * VLD + lhalf * 8];
    pf.h[0] = *(const v8h*)pp;
    pf.h[1] = *(const v8h*)(pp + 16);
    #pragma unroll
    for (int nt = 0; nt < 4; ++nt) {
      Frag16 bf;
      const _Float16* bp = &Vt[(nt * 16 + l16) * VLD + lhalf * 8];
      bf.h[0] = *(const v8h*)bp;
      bf.h[1] = *(const v8h*)(bp + 16);
      oacc[nt] = wmma16(pf, bf, oacc[nt]);
    }
    __syncthreads();
  }

  // ---- normalize + store (drop hd pad: only nt 0..2 cover hd<48)
  const int bb = bh >> 4, head = bh & 15;
  #pragma unroll
  for (int nt = 0; nt < 3; ++nt) {
    #pragma unroll
    for (int i = 0; i < 8; ++i) {
      int s = qrow0 + i + 8 * lhalf;
      int hd = nt * 16 + l16;
      float o = oacc[nt][i] * (1.0f / rs[i]);
      outh[(size_t)(bb * NSEQ + s) * DIMSZ + head * HDIM + hd] = (_Float16)o;
    }
  }
}

// ---------------------------------------------------------------- residual + LN
__global__ __launch_bounds__(256)
void add_ln_kernel(const float* __restrict__ X, const float* __restrict__ R,
                   const float* __restrict__ g, const float* __restrict__ be,
                   float* __restrict__ outF, _Float16* __restrict__ outH)
{
  __shared__ float red1[8], red2[8];
  const int row = blockIdx.x, tid = threadIdx.x;
  const float* xr = X + (size_t)row * DIMSZ;
  const float* rr = R + (size_t)row * DIMSZ;
  float v0[3]; float s = 0.0f;
  #pragma unroll
  for (int j = 0; j < 3; ++j) { int c = tid + 256 * j; v0[j] = xr[c] + rr[c]; s += v0[j]; }
  #pragma unroll
  for (int o = 16; o > 0; o >>= 1) s += __shfl_xor(s, o, 32);
  if ((tid & 31) == 0) red1[tid >> 5] = s;
  __syncthreads();
  float tot = 0.0f;
  #pragma unroll
  for (int w = 0; w < 8; ++w) tot += red1[w];
  float mean = tot * (1.0f / DIMSZ);
  float qs = 0.0f;
  #pragma unroll
  for (int j = 0; j < 3; ++j) { float d = v0[j] - mean; qs += d * d; }
  #pragma unroll
  for (int o = 16; o > 0; o >>= 1) qs += __shfl_xor(qs, o, 32);
  if ((tid & 31) == 0) red2[tid >> 5] = qs;
  __syncthreads();
  float tv = 0.0f;
  #pragma unroll
  for (int w = 0; w < 8; ++w) tv += red2[w];
  float inv = rsqrtf(tv * (1.0f / DIMSZ) + 1e-12f);
  #pragma unroll
  for (int j = 0; j < 3; ++j) {
    int c = tid + 256 * j;
    float y = (v0[j] - mean) * inv * g[c] + be[c];
    outF[(size_t)row * DIMSZ + c] = y;
    if (outH) outH[(size_t)row * DIMSZ + c] = (_Float16)y;
  }
}

// ---------------------------------------------------------------- launch
extern "C" void kernel_launch(void* const* d_in, const int* in_sizes, int n_in,
                              void* d_out, int out_size, void* d_ws, size_t ws_size,
                              hipStream_t stream) {
  const float* x      = (const float*)d_in[0];
  const float* w_qkv  = (const float*)d_in[1];
  const float* w_proj = (const float*)d_in[2];
  const float* b_proj = (const float*)d_in[3];
  const float* ln1_g  = (const float*)d_in[4];
  const float* ln1_b  = (const float*)d_in[5];
  const float* w1     = (const float*)d_in[6];
  const float* b1     = (const float*)d_in[7];
  const float* w2     = (const float*)d_in[8];
  const float* b2     = (const float*)d_in[9];
  const float* ln2_g  = (const float*)d_in[10];
  const float* ln2_b  = (const float*)d_in[11];
  float* out = (float*)d_out;

  char* ws = (char*)d_ws;
  size_t off = 0;
  auto alloc = [&](size_t bytes) -> char* {
    char* p = ws + off; off += (bytes + 255) & ~(size_t)255; return p;
  };
  const size_t QKVB = (size_t)NBATCH * NHEAD * NSEQ * HDPAD * 2;
  _Float16* xh     = (_Float16*)alloc((size_t)ROWS * DIMSZ * 2);
  _Float16* wqkvh  = (_Float16*)alloc((size_t)DIMSZ * 3 * DIMSZ * 2);
  _Float16* wprojh = (_Float16*)alloc((size_t)DIMSZ * DIMSZ * 2);
  _Float16* w1h    = (_Float16*)alloc((size_t)DIMSZ * HIDSZ * 2);
  _Float16* w2h    = (_Float16*)alloc((size_t)HIDSZ * DIMSZ * 2);
  _Float16* qhb    = (_Float16*)alloc(QKVB);
  _Float16* khb    = (_Float16*)alloc(QKVB);
  _Float16* vhb    = (_Float16*)alloc(QKVB);
  _Float16* attnh  = (_Float16*)alloc((size_t)ROWS * DIMSZ * 2);
  float*    tmp32  = (float*)   alloc((size_t)ROWS * DIMSZ * 4);
  float*    y1     = (float*)   alloc((size_t)ROWS * DIMSZ * 4);
  _Float16* y1h    = (_Float16*)alloc((size_t)ROWS * DIMSZ * 2);
  _Float16* hhb    = (_Float16*)alloc((size_t)ROWS * HIDSZ * 2);
  (void)ws_size; (void)in_sizes; (void)n_in; (void)out_size;

  auto cvt = [&](const float* src, _Float16* dst, int n) {
    cvt_f16<<<(n + 255) / 256, 256, 0, stream>>>(src, dst, n);
  };
  cvt(x,      xh,     ROWS * DIMSZ);
  cvt(w_qkv,  wqkvh,  DIMSZ * 3 * DIMSZ);
  cvt(w_proj, wprojh, DIMSZ * DIMSZ);
  cvt(w1,     w1h,    DIMSZ * HIDSZ);
  cvt(w2,     w2h,    HIDSZ * DIMSZ);

  // zero padded q/k/v (contiguous in ws) so hd 48..63 pads are 0
  (void)hipMemsetAsync(qhb, 0, 3 * QKVB, stream);

  const float qscale = 0.14433756729740643f;  // 1/sqrt(48)

  // qkv = x @ w_qkv  -> scatter q/k/v
  gemm_wmma<2><<<dim3((ROWS / 128) * (3 * DIMSZ / 128)), 256, 0, stream>>>(
      xh, wqkvh, nullptr, ROWS, 3 * DIMSZ, DIMSZ,
      nullptr, nullptr, qhb, khb, vhb, qscale);

  // attention
  attn_wmma<<<dim3(NBATCH * NHEAD * (NSEQ / 64)), 128, 0, stream>>>(
      qhb, khb, vhb, attnh);

  // proj
  gemm_wmma<0><<<dim3((ROWS / 128) * (DIMSZ / 128)), 256, 0, stream>>>(
      attnh, wprojh, b_proj, ROWS, DIMSZ, DIMSZ,
      tmp32, nullptr, nullptr, nullptr, nullptr, 0.0f);

  // x + proj -> LN1 -> y1 (f32) + y1h (f16)
  add_ln_kernel<<<dim3(ROWS), 256, 0, stream>>>(x, tmp32, ln1_g, ln1_b, y1, y1h);

  // h = gelu(y1 @ w1 + b1) -> f16
  gemm_wmma<1><<<dim3((ROWS / 128) * (HIDSZ / 128)), 256, 0, stream>>>(
      y1h, w1h, b1, ROWS, HIDSZ, DIMSZ,
      nullptr, hhb, nullptr, nullptr, nullptr, 0.0f);

  // mlp = h @ w2 + b2 -> f32
  gemm_wmma<0><<<dim3((ROWS / 128) * (DIMSZ / 128)), 256, 0, stream>>>(
      hhb, w2h, b2, ROWS, DIMSZ, HIDSZ,
      tmp32, nullptr, nullptr, nullptr, nullptr, 0.0f);

  // y1 + mlp -> LN2 -> out
  add_ln_kernel<<<dim3(ROWS), 256, 0, stream>>>(y1, tmp32, ln2_g, ln2_b, out, nullptr);
}
